// SparseLinearV_27573690040590
// MI455X (gfx1250) — matrix operand order repaired and checked by reading, not compiled, not run on hardware
//
#include <hip/hip_runtime.h>

#define IN_F   4096
#define OUT_F  4096
#define NCOL   4096

typedef __attribute__((ext_vector_type(16))) __bf16 v16bf;
typedef __attribute__((ext_vector_type(8)))  float  v8f;
typedef int v4i __attribute__((vector_size(16)));

union Frag {
    v16bf v;
    uint4 u[2];
    unsigned short h[16];
};

__device__ __forceinline__ unsigned short f2bf(float f) {
    union { float f; unsigned int u; } c; c.f = f;
    unsigned int u = c.u;
    u += 0x7FFFu + ((u >> 16) & 1u);   // round-to-nearest-even
    return (unsigned short)(u >> 16);
}

// ---------------- zero W (fp32) ----------------
__global__ void zero_f32_kernel(float* __restrict__ p, int n4) {
    int i = blockIdx.x * blockDim.x + threadIdx.x;
    if (i < n4) {
        float4 z; z.x = 0.f; z.y = 0.f; z.z = 0.f; z.w = 0.f;
        ((float4*)p)[i] = z;
    }
}

// ---------------- COO scatter with accumulation ----------------
__global__ void scatter_coo_kernel(const int* __restrict__ rows,
                                   const int* __restrict__ cols,
                                   const float* __restrict__ vals,
                                   float* __restrict__ W, int nnz) {
    int i = blockIdx.x * blockDim.x + threadIdx.x;
    if (i < nnz) {
        long idx = (long)rows[i] * OUT_F + cols[i];
        atomicAdd(W + idx, vals[i]);
    }
}

// ---------------- fp32 -> bf16 convert (vectorized x4, no transpose) ------
__global__ void cvt_f32_bf16_kernel(const float* __restrict__ src,
                                    unsigned short* __restrict__ dst, int n4) {
    int i = blockIdx.x * blockDim.x + threadIdx.x;
    if (i < n4) {
        float4 v = ((const float4*)src)[i];
        uint2 o;
        o.x = (unsigned int)f2bf(v.x) | ((unsigned int)f2bf(v.y) << 16);
        o.y = (unsigned int)f2bf(v.z) | ((unsigned int)f2bf(v.w) << 16);
        ((uint2*)dst)[i] = o;
    }
}

// ---------------- fp32 -> bf16 transposing convert -----------------------
// src: (R x C) fp32 row-major; dst: (C x R) bf16 row-major
__global__ void transpose_cvt_kernel(const float* __restrict__ src,
                                     unsigned short* __restrict__ dst) {
    const int R = 4096, C = 4096;
    __shared__ unsigned short tile[32][33];
    const int gx = blockIdx.x * 32;   // source column
    const int gy = blockIdx.y * 32;   // source row
    const int tx = threadIdx.x, ty = threadIdx.y;
    #pragma unroll
    for (int j = 0; j < 32; j += 8)
        tile[ty + j][tx] = f2bf(src[(long)(gy + ty + j) * C + gx + tx]);
    __syncthreads();
    #pragma unroll
    for (int j = 0; j < 32; j += 8)
        dst[(long)(gx + ty + j) * R + gy + tx] = tile[tx][ty + j];
}

// ---------------- bf16 WMMA GEMM: C = A * Bt^T + bias --------------------
// A : (4096 x 4096) bf16 row-major, K contiguous        [= W]
// Bt: (4096 x 4096) bf16 row-major, K contiguous        [= x^T]
// C : fp32 (M x N); bias added along columns (N = out_features)
#define BM 256
#define BN 128
#define BK 64
#define PADK 8   // row stride 72 elems = 144B: 16B-aligned, bank-spread
#define LDSTRIDE (BK + PADK)

#if defined(__has_builtin)
#if __has_builtin(__builtin_amdgcn_global_load_async_to_lds_b128)
#define HAVE_ASYNC_LDS 1
#endif
#endif

__device__ __forceinline__ void stage_b128(const unsigned short* g,
                                           unsigned short* l) {
#ifdef HAVE_ASYNC_LDS
    __builtin_amdgcn_global_load_async_to_lds_b128(
        (__attribute__((address_space(1))) v4i*)g,
        (__attribute__((address_space(3))) v4i*)l,
        0, 0);
#else
    *(uint4*)l = *(const uint4*)g;
#endif
}

template <int N>
__device__ __forceinline__ void stage_wait() {
#ifdef HAVE_ASYNC_LDS
    __builtin_amdgcn_s_wait_asynccnt(N);
#endif
}

__device__ __forceinline__ void compute_tile(
    const unsigned short (*__restrict__ Asb)[LDSTRIDE],
    const unsigned short (*__restrict__ Bsb)[LDSTRIDE],
    v8f acc[4][4], int wm, int wn, int lr, int kh, int kb) {
    #pragma unroll
    for (int kk = 0; kk < BK; kk += 32) {
        // A fragments: lane lr = row; K = {kh..kh+7, kh+16..kh+23}
        Frag af[4];
        #pragma unroll
        for (int mi = 0; mi < 4; ++mi) {
            const unsigned short* p = &Asb[wm + mi * 16 + lr][kk + kh];
            af[mi].u[0] = *(const uint4*)p;
            af[mi].u[1] = *(const uint4*)(p + 16);
        }
        // B fragments: lane lr = column; K = {kh..kh+7, kh+16..kh+23}
        Frag bfr[4];
        #pragma unroll
        for (int ni = 0; ni < 4; ++ni) {
            const unsigned short* q = &Bsb[wn + ni * 16 + lr][kk + kh];
            bfr[ni].u[0] = *(const uint4*)q;
            bfr[ni].u[1] = *(const uint4*)(q + 16);
        }
        #pragma unroll
        for (int mi = 0; mi < 4; ++mi)
            #pragma unroll
            for (int ni = 0; ni < 4; ++ni)
                acc[mi][ni] = __builtin_amdgcn_wmma_f32_16x16x32_bf16(
                    false, af[mi].v, false, bfr[ni].v,
                    (short)0, acc[mi][ni], false, false);
    }
}

__global__ __launch_bounds__(256)
void wmma_gemm_bf16_kernel(const unsigned short* __restrict__ A,
                           const unsigned short* __restrict__ Bt,
                           const float* __restrict__ bias,
                           float* __restrict__ C) {
    const int K = 4096, N = 4096;
    __shared__ unsigned short As[2][BM][LDSTRIDE];   // 2*256*72*2 = 73.7 KB
    __shared__ unsigned short Bs[2][BN][LDSTRIDE];   // 2*128*72*2 = 36.9 KB

    const int tid  = threadIdx.x;
    const int lane = tid & 31;
    const int wave = tid >> 5;

    const int m0 = blockIdx.y * BM;
    const int n0 = blockIdx.x * BN;

    // 8 waves: 4 along M, 2 along N; each wave owns a 64x64 C tile
    const int wm = (wave & 3) * 64;
    const int wn = (wave >> 2) * 64;

    const int lr = lane & 15;          // row (A) / col (B) within a 16-tile
    const int kh = (lane >> 4) * 8;    // A: K half-offset {0,8}
    const int kb = (lane >> 4) * 16;   // unused; B also uses kh {0,8}

    // Per-thread staging coordinates (b128 chunks)
    const int sr = tid >> 3;           // 0..31 (row step 32 per i)
    const int sc = (tid & 7) * 8;      // K chunk within BK

    v8f acc[4][4];
    #pragma unroll
    for (int mi = 0; mi < 4; ++mi)
        #pragma unroll
        for (int ni = 0; ni < 4; ++ni)
            #pragma unroll
            for (int e = 0; e < 8; ++e) acc[mi][ni][e] = 0.f;

    // stage one k-block into buffer `buf`: 12 async ops per thread/wave
    auto stage = [&](int buf, int k0) {
        #pragma unroll
        for (int i = 0; i < 8; ++i) {   // A: 256 rows x 64 K
            int r = sr + i * 32;
            stage_b128(A + (long)(m0 + r) * K + k0 + sc, &As[buf][r][sc]);
        }
        #pragma unroll
        for (int i = 0; i < 4; ++i) {   // B: 128 rows x 64 K
            int r = sr + i * 32;
            stage_b128(Bt + (long)(n0 + r) * K + k0 + sc, &Bs[buf][r][sc]);
        }
    };

    // ---- software pipeline: copy(k+1) overlaps compute(k) ----
    stage(0, 0);                        // asynccnt = 12
    int buf = 0;
    for (int k0 = 0; k0 < K - BK; k0 += BK) {
        stage(buf ^ 1, k0 + BK);        // asynccnt = 24
        stage_wait<12>();               // tile k done (async loads retire in order)
        __syncthreads();
        compute_tile(As[buf], Bs[buf], acc, wm, wn, lr, kh, kb);
        __syncthreads();                // all waves done before buf is re-staged
        buf ^= 1;
    }
    stage_wait<0>();
    __syncthreads();
    compute_tile(As[buf], Bs[buf], acc, wm, wn, lr, kh, kb);

    // Epilogue: C layout => row = vreg + 8*(lane>=16), col = lane&15
    const int rbase = (lane >> 4) * 8;
    const int cN    = lane & 15;
    #pragma unroll
    for (int ni = 0; ni < 4; ++ni) {
        int col = n0 + wn + ni * 16 + cN;
        float b = bias[col];
        #pragma unroll
        for (int mi = 0; mi < 4; ++mi) {
            int row0 = m0 + wm + mi * 16 + rbase;
            #pragma unroll
            for (int v = 0; v < 8; ++v) {
                C[(long)(row0 + v) * N + col] = acc[mi][ni][v] + b;
            }
        }
    }
}

extern "C" void kernel_launch(void* const* d_in, const int* in_sizes, int n_in,
                              void* d_out, int out_size, void* d_ws, size_t ws_size,
                              hipStream_t stream) {
    const float* x    = (const float*)d_in[0];
    const int*   rows = (const int*)d_in[1];
    const int*   cols = (const int*)d_in[2];
    const float* vals = (const float*)d_in[3];
    const float* bias = (const float*)d_in[4];
    float* out = (float*)d_out;
    const int nnz = in_sizes[1];

    char* ws = (char*)d_ws;
    const size_t nelem = (size_t)IN_F * OUT_F;          // 16,777,216
    float*          Wf  = (float*)ws;                   // 64 MB fp32 dense W
    unsigned short* Wbf = (unsigned short*)(ws + nelem * 4);            // 32 MB
    unsigned short* Xt  = (unsigned short*)(ws + nelem * 4 + nelem * 2);// 32 MB (x^T bf16)

    const int n4  = (int)(nelem / 4);       // 4,194,304
    const int blk = 256;

    zero_f32_kernel<<<n4 / blk, blk, 0, stream>>>(Wf, n4);
    scatter_coo_kernel<<<(nnz + blk - 1) / blk, blk, 0, stream>>>(rows, cols, vals, Wf, nnz);
    cvt_f32_bf16_kernel<<<n4 / blk, blk, 0, stream>>>(Wf, Wbf, n4);

    dim3 tgrid(NCOL / 32, IN_F / 32);
    dim3 tblk(32, 8);
    transpose_cvt_kernel<<<tgrid, tblk, 0, stream>>>(x, Xt);

    dim3 grid(NCOL / BN, IN_F / BM);
    wmma_gemm_bf16_kernel<<<grid, 256, 0, stream>>>(Wbf, Xt, bias, out);
}
